// Net_42786464203126
// MI455X (gfx1250) — matrix operand order repaired and checked by reading, not compiled, not run on hardware
//
#include <hip/hip_runtime.h>
#include <cmath>

// ---------------------------------------------------------------------------
// SplineConv (K=5^3 kernels, degree-1 basis, F_IN=1, F_OUT=32) + ELU +
// sparse voxel max pool, for MI455X (gfx1250, wave32, WMMA).
//
// Edge message einsum is recast as dense GEMM tiles so it runs on the matrix
// pipe: per 16-edge tile  [16 x 128]f16 (sparse-expanded basis*x) times
// [128 x 32]f16 (padded W^T) -> [16 x 32]f32 via v_wmma_f32_16x16x32_f16.
// Aggregation uses f32 atomics into an L2-resident 8MB array; pooling uses a
// monotone uint mapping + atomicMax(u32).
// ---------------------------------------------------------------------------

typedef _Float16 f16;
typedef __attribute__((ext_vector_type(16))) _Float16 v16h;
typedef __attribute__((ext_vector_type(8)))  float    v8f;

struct alignas(16) U4   { unsigned x, y, z, w; };
struct alignas(32) Frag { U4 lo, hi; };

constexpr int NN    = 65536;     // nodes
constexpr int EE    = 1048576;   // edges
constexpr int FOUT  = 32;
constexpr int NVOX  = 32768;     // 32^3
constexpr int KPAD  = 128;       // 125 kernels padded to 128
constexpr int WPB   = 4;         // waves per block (edge kernel) -> 32KB LDS

__device__ __forceinline__ unsigned fkey(float f) {
  unsigned b = __float_as_uint(f);
  return (b & 0x80000000u) ? ~b : (b | 0x80000000u);   // monotone float->u32
}
__device__ __forceinline__ float fdec(unsigned k) {
  return __uint_as_float((k & 0x80000000u) ? (k & 0x7FFFFFFFu) : ~k);
}

// Zero agg/deg, reset pooled keys in d_out, build padded W^T in f16.
__global__ __launch_bounds__(256)
void init_kernel(float* __restrict__ agg, float* __restrict__ deg,
                 f16* __restrict__ Wt, const float* __restrict__ W,
                 unsigned* __restrict__ outk) {
  int i = blockIdx.x * 256 + threadIdx.x;        // grid covers N*FOUT = 2M
  agg[i] = 0.0f;
  if (i < NVOX * FOUT) outk[i] = 0u;             // key sentinel (== -inf)
  if (i < NN) deg[i] = 0.0f;
  if (i < FOUT * KPAD) {                         // Wt[o][k] = W[k][0][o]
    int o = i >> 7, k = i & 127;
    float v = (k < 125) ? W[k * FOUT + o] : 0.0f;
    Wt[i] = (f16)v;
  }
}

// One wave = 32 edges = two 16x32 message tiles via WMMA.
__global__ __launch_bounds__(32 * WPB)
void edge_kernel(const float* __restrict__ x, const int* __restrict__ ei,
                 const float* __restrict__ pseudo, const f16* __restrict__ Wt,
                 float* __restrict__ agg, float* __restrict__ deg) {
  __shared__ f16 Alds[WPB * 32 * KPAD];          // 32KB: per-wave 32x128 A tile
  __shared__ int dstlds[WPB * 32];

  const int lane  = threadIdx.x & 31;
  const int wave  = threadIdx.x >> 5;
  const int half  = lane >> 4;                   // lane group (0: lanes 0-15)
  const int wbase = wave * 32 * KPAD;
  const int e     = (blockIdx.x * WPB + wave) * 32 + lane;

  // --- preload B fragments (shared W^T, 8 frags x 32B/lane, L2-hot) --------
  // B k order used here (contiguous, same as A): lanes 0-15 hold col=lane K 0..15,
  // lanes 16-31 hold col=lane-16 K 16..31 (contiguous in Wt rows).
  v16h Bf[4][2];
#pragma unroll
  for (int kk = 0; kk < 4; ++kk)
#pragma unroll
    for (int h = 0; h < 2; ++h) {
      int col = (lane & 15) + h * 16;
      int ko  = kk * 32 + half * 16;
      const U4* p = (const U4*)(Wt + col * KPAD + ko);
      Frag fb; fb.lo = p[0]; fb.hi = p[1];
      Bf[kk][h] = __builtin_bit_cast(v16h, fb);
    }

  // --- per-edge degree-1 B-spline basis ------------------------------------
  const int   src = ei[e];
  const int   dst = ei[EE + e];
  const float xv  = x[src];
  float p0 = pseudo[3 * e], p1 = pseudo[3 * e + 1], p2 = pseudo[3 * e + 2];
  float v0 = p0 * 4.0f, v1 = p1 * 4.0f, v2 = p2 * 4.0f;
  int  i0 = min(max((int)floorf(v0), 0), 3);
  int  i1 = min(max((int)floorf(v1), 0), 3);
  int  i2 = min(max((int)floorf(v2), 0), 3);
  float f0 = v0 - (float)i0, f1 = v1 - (float)i1, f2 = v2 - (float)i2;
  int  base = i0 + 5 * i1 + 25 * i2;             // max 93; +31 < 125

  // zero own 256B A row, then scatter the 8 weighted basis values
  U4 z; z.x = z.y = z.z = z.w = 0u;
  U4* rp = (U4*)(Alds + wbase + lane * KPAD);
#pragma unroll
  for (int i = 0; i < 16; ++i) rp[i] = z;
  f16* row = Alds + wbase + lane * KPAD;
#pragma unroll
  for (int b = 0; b < 8; ++b) {
    float w = ((b & 1) ? f0 : 1.0f - f0) * ((b & 2) ? f1 : 1.0f - f1) *
              ((b & 4) ? f2 : 1.0f - f2) * xv;
    int k = base + (b & 1) + ((b >> 1) & 1) * 5 + ((b >> 2) & 1) * 25;
    row[k] = (f16)w;
  }
  dstlds[wave * 32 + lane] = dst;
  atomicAdd(&deg[dst], 1.0f);
  __syncthreads();                               // A tiles visible wave-wide

  // --- two 16-edge tiles: [16x128]x[128x32] -> [16x32] f32 ----------------
#pragma unroll
  for (int t = 0; t < 2; ++t) {
    v8f c0 = {}; v8f c1 = {};
#pragma unroll
    for (int kk = 0; kk < 4; ++kk) {
      // A k order used here (contiguous, same as B): lane<16 row=lane K{0..15},
      // lane>=16 row=lane-16 K{16..31}.
      int arow = t * 16 + (lane & 15);
      int ko   = kk * 32 + half * 16;
      const f16* ap = Alds + wbase + arow * KPAD + ko;
      Frag fa; fa.lo = *(const U4*)ap; fa.hi = *(const U4*)(ap + 8);
      v16h a = __builtin_bit_cast(v16h, fa);
      c0 = __builtin_amdgcn_wmma_f32_16x16x32_f16(false, a, false, Bf[kk][0],
                                                  (short)0, c0, false, false);
      c1 = __builtin_amdgcn_wmma_f32_16x16x32_f16(false, a, false, Bf[kk][1],
                                                  (short)0, c1, false, false);
    }
    // D layout: lane holds col=(lane&15)+h*16, VGPR r = row r (+8 for hi lanes)
    int col   = lane & 15;
    int rbase = t * 16 + half * 8;
#pragma unroll
    for (int r = 0; r < 8; ++r) {
      int d = dstlds[wave * 32 + rbase + r];
      atomicAdd(&agg[d * FOUT + col],       c0[r]);
      atomicAdd(&agg[d * FOUT + 16 + col],  c1[r]);
    }
  }
}

// mean aggregation + root weight + bias + ELU, then voxel scatter-max (keys).
__global__ __launch_bounds__(256)
void node_kernel(const float* __restrict__ agg, const float* __restrict__ deg,
                 const float* __restrict__ x, const float* __restrict__ Wroot,
                 const float* __restrict__ bias, const float* __restrict__ pos,
                 unsigned* __restrict__ outk) {
  int i = blockIdx.x * 256 + threadIdx.x;        // grid covers N*FOUT
  int n = i >> 5, o = i & 31;                    // one wave per node
  float val = agg[i] / fmaxf(deg[n], 1.0f) + x[n] * Wroot[o] + bias[o];
  float h = (val > 0.0f) ? val : expm1f(val);    // ELU(alpha=1)
  float q0 = pos[3 * n], q1 = pos[3 * n + 1], q2 = pos[3 * n + 2];
  int g0 = min(max((int)floorf(q0 * 32.0f), 0), 31);
  int g1 = min(max((int)floorf(q1 * 32.0f), 0), 31);
  int g2 = min(max((int)floorf(q2 * 32.0f), 0), 31);
  int vidx = g0 + 32 * g1 + 1024 * g2;
  atomicMax(&outk[vidx * FOUT + o], fkey(h));
}

// decode keys in place; untouched voxels (key==0 sentinel) -> 0.0
__global__ __launch_bounds__(256)
void fin_kernel(unsigned* __restrict__ outk) {
  int i = blockIdx.x * 256 + threadIdx.x;        // grid covers NVOX*FOUT
  unsigned k = outk[i];
  float f = (k == 0u) ? 0.0f : fdec(k);
  ((float*)outk)[i] = f;
}

extern "C" void kernel_launch(void* const* d_in, const int* in_sizes, int n_in,
                              void* d_out, int out_size, void* d_ws, size_t ws_size,
                              hipStream_t stream) {
  const float* x      = (const float*)d_in[0];
  const int*   ei     = (const int*)  d_in[1];   // [2,E]: src row, dst row
  const float* pseudo = (const float*)d_in[2];
  const float* pos    = (const float*)d_in[3];
  const float* W      = (const float*)d_in[4];   // [125,1,32]
  const float* Wroot  = (const float*)d_in[5];   // [1,32]
  const float* bias   = (const float*)d_in[6];   // [32]

  char*  ws  = (char*)d_ws;                      // ~8.3MB used
  float* agg = (float*)ws;                                           // 8MB
  float* deg = (float*)(ws + (size_t)NN * FOUT * 4);                 // 256KB
  f16*   Wt  = (f16*)  (ws + (size_t)NN * FOUT * 4 + (size_t)NN * 4);// 8KB
  unsigned* outk = (unsigned*)d_out;             // pooled keys live in d_out

  init_kernel<<<NN * FOUT / 256, 256, 0, stream>>>(agg, deg, Wt, W, outk);
  edge_kernel<<<EE / (32 * WPB), 32 * WPB, 0, stream>>>(x, ei, pseudo, Wt, agg, deg);
  node_kernel<<<NN * FOUT / 256, 256, 0, stream>>>(agg, deg, x, Wroot, bias, pos, outk);
  fin_kernel<<<NVOX * FOUT / 256, 256, 0, stream>>>(outk);
}
